// AttentionDecoder_7911329759250
// MI455X (gfx1250) — compile-verified, matrix operand order checked by reading, not run on hardware
//
#include <hip/hip_runtime.h>
#include <hip/hip_bf16.h>

// ---------------------------------------------------------------------------
// Problem constants
// ---------------------------------------------------------------------------
#define Vv 32000
#define Ee 512
#define Hh 512
#define Bb 64
#define Ss 128
#define G3H 1536   // 3*H

// Workspace layout (bytes, 256-aligned)
#define OFF_WIH   ((size_t)0)                    // [1536][512] bf16 (ushort)
#define OFF_WHH   ((size_t)0x180000)             // [1536][512] bf16
#define OFF_ENC   ((size_t)0x300000)             // [64][128][512] bf16
#define OFF_ENCT  ((size_t)0xB00000)             // [64][512][128] bf16
#define OFF_GI    ((size_t)0x1300000)            // swizzled [512 mtile][96 ntile][32 lane][8 r] fp32
#define OFF_HBF   ((size_t)0x4300000)            // [64][128][512] bf16
// total = 0x4B00000 = ~78.6 MB

typedef __attribute__((ext_vector_type(16))) __bf16 v16bf;
typedef __attribute__((ext_vector_type(8)))  float  v8f;

union FragBF {
    v16bf v;
    unsigned short u[16];
    uint4 q[2];
};

union F8 {
    float4 q[2];
    float  e[8];
};

static __device__ inline v8f vz8() {
    v8f z;
#pragma unroll
    for (int i = 0; i < 8; ++i) z[i] = 0.0f;
    return z;
}

// float -> bf16 (RNE) as raw ushort
static __device__ inline unsigned short f2bf(float f) {
    union { float f; unsigned u; } x; x.f = f;
    return (unsigned short)((x.u + 0x7FFFu + ((x.u >> 16) & 1u)) >> 16);
}

// 8 consecutive floats -> 8 packed bf16 in a uint4
static __device__ inline uint4 cvt8(const float* __restrict__ src) {
    float4 f0 = *(const float4*)src;
    float4 f1 = *(const float4*)(src + 4);
    uint4 d;
    d.x = (unsigned)f2bf(f0.x) | ((unsigned)f2bf(f0.y) << 16);
    d.y = (unsigned)f2bf(f0.z) | ((unsigned)f2bf(f0.w) << 16);
    d.z = (unsigned)f2bf(f1.x) | ((unsigned)f2bf(f1.y) << 16);
    d.w = (unsigned)f2bf(f1.z) | ((unsigned)f2bf(f1.w) << 16);
    return d;
}

static __device__ inline v8f wmma_bf16(const FragBF& a, const FragBF& b, v8f c) {
    return __builtin_amdgcn_wmma_f32_16x16x32_bf16(false, a.v, false, b.v,
                                                   (short)0, c, false, false);
}

// ---------------------------------------------------------------------------
// Kernel 0: precision prep — bf16 copies of W_ih, W_hh, enc, enc^T (packed)
// ---------------------------------------------------------------------------
__global__ void prep_kernel(const float* __restrict__ Wih,
                            const float* __restrict__ Whh,
                            const float* __restrict__ enc,
                            unsigned short* __restrict__ WihB,
                            unsigned short* __restrict__ WhhB,
                            unsigned short* __restrict__ encB,
                            unsigned short* __restrict__ encT) {
    const long NW8 = ((long)G3H * Ee) / 8;       //  98304 uint4 chunks
    const long NE8 = ((long)Bb * Ss * Hh) / 8;   // 524288 uint4 chunks
    const long gid = (long)blockIdx.x * blockDim.x + threadIdx.x;
    const long stride = (long)gridDim.x * blockDim.x;

    for (long i = gid; i < NW8; i += stride)
        ((uint4*)WihB)[i] = cvt8(Wih + i * 8);
    for (long i = gid; i < NW8; i += stride)
        ((uint4*)WhhB)[i] = cvt8(Whh + i * 8);
    for (long i = gid; i < NE8; i += stride)
        ((uint4*)encB)[i] = cvt8(enc + i * 8);

    // encT[b][h][s] = enc[b][s][h]; pack 8 consecutive s per uint4 store
    for (long i = gid; i < NE8; i += stride) {
        long j   = i * 8;
        long b   = j >> 16;          // /65536
        long rem = j & 65535;
        long h   = rem >> 7;
        long s0  = rem & 127;
        float t[8];
#pragma unroll
        for (int k = 0; k < 8; ++k)
            t[k] = enc[((b << 7) + s0 + k) * Hh + h];
        uint4 d;
        d.x = (unsigned)f2bf(t[0]) | ((unsigned)f2bf(t[1]) << 16);
        d.y = (unsigned)f2bf(t[2]) | ((unsigned)f2bf(t[3]) << 16);
        d.z = (unsigned)f2bf(t[4]) | ((unsigned)f2bf(t[5]) << 16);
        d.w = (unsigned)f2bf(t[6]) | ((unsigned)f2bf(t[7]) << 16);
        ((uint4*)encT)[i] = d;
    }
}

// ---------------------------------------------------------------------------
// Kernel 1: GI = emb[token] @ W_ih^T + b_ih   (M=8192,N=1536,K=512)
// grid (512 M-tiles, 6 N-groups of 256), block 256 (8 waves)
// Output in WMMA C-fragment-native layout [mtile][ntile][lane][r].
// ---------------------------------------------------------------------------
__global__ void gi_gemm_kernel(const int* __restrict__ tok,
                               const float* __restrict__ emb,
                               const unsigned short* __restrict__ WihB,
                               const float* __restrict__ b_ih,
                               float* __restrict__ GI) {
    __shared__ __align__(16) unsigned short As[16 * 48];
    __shared__ __align__(16) unsigned short Bs[256 * 48];
    __shared__ int tokS[16];

    const int tid   = threadIdx.x;
    const int mtile = blockIdx.x;            // M = s*64 + b; mtile = s*4 + btile
    const int n0    = blockIdx.y * 256;
    const int s     = mtile >> 2;
    const int b0    = (mtile & 3) << 4;

    if (tid < 16) tokS[tid] = tok[(b0 + tid) * Ss + s];
    __syncthreads();

    const int wv   = tid >> 5;
    const int lane = tid & 31;
    const int m    = lane & 15;
    const int kh   = lane >> 4;

    v8f acc0 = vz8(), acc1 = vz8();

    for (int c = 0; c < 16; ++c) {
        const int k0 = c << 5;
        // stage A tile (16x32): 64 threads, one packed b128 each
        if (tid < 64) {
            int row = tid >> 2, seg = tid & 3;
            *(uint4*)&As[row * 48 + seg * 8] =
                cvt8(&emb[(size_t)tokS[row] * Ee + k0 + seg * 8]);
        }
        // stage B tile (256x32 from bf16 W_ih), 16B chunks
#pragma unroll
        for (int p = 0; p < 4; ++p) {
            int cc  = tid + p * 256;
            int row = cc >> 2, seg = cc & 3;
            *(uint4*)&Bs[row * 48 + seg * 8] =
                *(const uint4*)&WihB[(size_t)(n0 + row) * Ee + k0 + seg * 8];
        }
        __syncthreads();

        FragBF a;
        a.q[0] = *(const uint4*)&As[m * 48 + kh * 8];
        a.q[1] = *(const uint4*)&As[m * 48 + 16 + kh * 8];

        FragBF bf0, bf1;
        int r0 = (wv * 2) * 16 + m;
        int r1 = (wv * 2 + 1) * 16 + m;
        bf0.q[0] = *(const uint4*)&Bs[r0 * 48 + kh * 8];
        bf0.q[1] = *(const uint4*)&Bs[r0 * 48 + 16 + kh * 8];
        bf1.q[0] = *(const uint4*)&Bs[r1 * 48 + kh * 8];
        bf1.q[1] = *(const uint4*)&Bs[r1 * 48 + 16 + kh * 8];

        acc0 = wmma_bf16(a, bf0, acc0);
        acc1 = wmma_bf16(a, bf1, acc1);
        __syncthreads();
    }

    // epilogue: + b_ih, store in fragment-native swizzled layout (coalesced)
#pragma unroll
    for (int i = 0; i < 2; ++i) {
        const v8f acc = i ? acc1 : acc0;
        const int ntile = blockIdx.y * 16 + wv * 2 + i;      // 0..95
        const float bias = b_ih[ntile * 16 + m];
        float* dst = &GI[(((size_t)mtile * 96 + ntile) * 32 + lane) * 8];
        F8 o;
#pragma unroll
        for (int r = 0; r < 8; ++r) o.e[r] = acc[r] + bias;
        *(float4*)dst       = o.q[0];
        *(float4*)(dst + 4) = o.q[1];
    }
}

// ---------------------------------------------------------------------------
// Kernel 2: sequential GRU recurrence. grid 4 (batch tiles of 16), block 512.
// Writes h_t (fp32) to out[b][t][0:512] and bf16 copy to Hbf[b][t][:].
// ---------------------------------------------------------------------------
__global__ void recurrence_kernel(const float* __restrict__ h0,
                                  const unsigned short* __restrict__ WhhB,
                                  const float* __restrict__ GI,
                                  const float* __restrict__ b_hh,
                                  unsigned short* __restrict__ Hbf,
                                  float* __restrict__ out) {
    __shared__ float hf[16 * 512];
    __shared__ __align__(16) unsigned short hb[16 * 520];

    const int tid  = threadIdx.x;
    const int b0   = blockIdx.x * 16;
    const int wv   = tid >> 5;
    const int lane = tid & 31;
    const int m    = lane & 15;
    const int kh   = lane >> 4;

    for (int e = tid; e < 16 * 512; e += 512) {
        int row = e >> 9, col = e & 511;
        float v = h0[(size_t)(b0 + row) * Hh + col];
        hf[row * 512 + col] = v;
        hb[row * 520 + col] = f2bf(v);
    }
    __syncthreads();

    const int jt0 = wv * 2;   // this wave's two hidden-column tiles

    for (int t = 0; t < Ss; ++t) {
        v8f acc[2][3];
#pragma unroll
        for (int ji = 0; ji < 2; ++ji)
#pragma unroll
            for (int g = 0; g < 3; ++g) acc[ji][g] = vz8();

        for (int c = 0; c < 16; ++c) {
            const int k0 = c << 5;
            FragBF a;
            a.q[0] = *(const uint4*)&hb[m * 520 + k0 + kh * 8];
            a.q[1] = *(const uint4*)&hb[m * 520 + k0 + 16 + kh * 8];
            if (c < 15)
                __builtin_prefetch(&WhhB[(size_t)((jt0 << 4) + m) * Hh + k0 + 32], 0, 1);
#pragma unroll
            for (int ji = 0; ji < 2; ++ji) {
#pragma unroll
                for (int g = 0; g < 3; ++g) {
                    const int n = g * Hh + (jt0 + ji) * 16 + m;  // W_hh row
                    const unsigned short* src = &WhhB[(size_t)n * Hh + k0 + kh * 8];
                    FragBF bf;
                    bf.q[0] = *(const uint4*)src;
                    bf.q[1] = *(const uint4*)(src + 16);
                    acc[ji][g] = wmma_bf16(a, bf, acc[ji][g]);
                }
            }
        }

        // gates (fp32); GI is in fragment-native layout -> contiguous b128 loads
        const size_t mt_c = (size_t)(t * 4 + blockIdx.x);
        float hnew[2][8];
#pragma unroll
        for (int ji = 0; ji < 2; ++ji) {
            const int jt   = jt0 + ji;
            const int ncol = jt * 16 + m;
            const float bhr = b_hh[ncol];
            const float bhz = b_hh[Hh + ncol];
            const float bhn = b_hh[2 * Hh + ncol];

            const float* gr_p = &GI[((mt_c * 96 + (jt))      * 32 + lane) * 8];
            const float* gz_p = &GI[((mt_c * 96 + (32 + jt)) * 32 + lane) * 8];
            const float* gn_p = &GI[((mt_c * 96 + (64 + jt)) * 32 + lane) * 8];
            F8 gR, gZ, gN;
            gR.q[0] = *(const float4*)gr_p; gR.q[1] = *(const float4*)(gr_p + 4);
            gZ.q[0] = *(const float4*)gz_p; gZ.q[1] = *(const float4*)(gz_p + 4);
            gN.q[0] = *(const float4*)gn_p; gN.q[1] = *(const float4*)(gn_p + 4);

#pragma unroll
            for (int r = 0; r < 8; ++r) {
                const int mrow = r + 8 * kh;
                float gr = gR.e[r] + acc[ji][0][r] + bhr;
                float gz = gZ.e[r] + acc[ji][1][r] + bhz;
                float rr = 1.0f / (1.0f + __expf(-gr));
                float zz = 1.0f / (1.0f + __expf(-gz));
                float nn = tanhf(gN.e[r] + rr * (acc[ji][2][r] + bhn));
                float ho = hf[mrow * 512 + ncol];
                hnew[ji][r] = (1.0f - zz) * nn + zz * ho;
            }
        }

        __syncthreads();   // all waves done reading hb/hf for this step
#pragma unroll
        for (int ji = 0; ji < 2; ++ji) {
            const int ncol = (jt0 + ji) * 16 + m;
#pragma unroll
            for (int r = 0; r < 8; ++r) {
                const int mrow = r + 8 * kh;
                const float v = hnew[ji][r];
                hf[mrow * 512 + ncol] = v;
                hb[mrow * 520 + ncol] = f2bf(v);
                const int gb = b0 + mrow;
                out[((size_t)gb * Ss + t) * 1024 + ncol] = v;
                Hbf[((size_t)gb * Ss + t) * Hh + ncol]   = f2bf(v);
            }
        }
        __syncthreads();   // writes visible before next step's reads
    }
}

// ---------------------------------------------------------------------------
// Kernel 3: attention. grid (8 t-tiles, 64 batches), block 256 (8 waves).
// scores = Hbf[b] @ enc[b]^T -> softmax -> context = scores @ enc[b]
// ---------------------------------------------------------------------------
__global__ void attention_kernel(const unsigned short* __restrict__ Hbf,
                                 const unsigned short* __restrict__ encB,
                                 const unsigned short* __restrict__ encT,
                                 float* __restrict__ out) {
    __shared__ float sc[16 * 132];   // 16 rows (t) x 128 cols (s), stride 132

    const int tid  = threadIdx.x;
    const int mt   = blockIdx.x;
    const int b    = blockIdx.y;
    const int wv   = tid >> 5;
    const int lane = tid & 31;
    const int m    = lane & 15;
    const int kh   = lane >> 4;

    // ---- phase 1: scores tile (M=16 t-rows, N=128 s, K=512) ----
    {
        v8f acc = vz8();
        const size_t arow = ((size_t)b * Ss + mt * 16 + m) * Hh;
        const size_t brow = ((size_t)b * Ss + wv * 16 + m) * Hh;
        for (int c = 0; c < 16; ++c) {
            const int k0 = c << 5;
            FragBF a, bf;
            a.q[0]  = *(const uint4*)&Hbf[arow + k0 + kh * 8];
            a.q[1]  = *(const uint4*)&Hbf[arow + k0 + 16 + kh * 8];
            bf.q[0] = *(const uint4*)&encB[brow + k0 + kh * 8];
            bf.q[1] = *(const uint4*)&encB[brow + k0 + 16 + kh * 8];
            acc = wmma_bf16(a, bf, acc);
        }
#pragma unroll
        for (int r = 0; r < 8; ++r) {
            int srow = r + 8 * kh;
            sc[srow * 132 + wv * 16 + m] = acc[r];
        }
    }
    __syncthreads();

    // ---- softmax over s (one row per thread, 16 rows) ----
    if (tid < 16) {
        float* row = &sc[tid * 132];
        float mx = -1e30f;
        for (int s = 0; s < Ss; ++s) mx = fmaxf(mx, row[s]);
        float sum = 0.0f;
        for (int s = 0; s < Ss; ++s) { float e = __expf(row[s] - mx); row[s] = e; sum += e; }
        float inv = 1.0f / sum;
        for (int s = 0; s < Ss; ++s) row[s] *= inv;
    }
    __syncthreads();

    // ---- phase 2: context (M=16 t-rows, N=512 h, K=128) ----
    FragBF afr[4];
#pragma unroll
    for (int c = 0; c < 4; ++c) {
        const int k0 = c << 5;
#pragma unroll
        for (int e = 0; e < 16; ++e) {
            int kk = k0 + (e >> 3) * 16 + kh * 8 + (e & 7);
            afr[c].u[e] = f2bf(sc[m * 132 + kk]);
        }
    }

#pragma unroll
    for (int nt = 0; nt < 4; ++nt) {
        const int hcol = (wv * 4 + nt) * 16 + m;
        const size_t brow = ((size_t)b * Hh + hcol) * Ss;
        v8f acc = vz8();
#pragma unroll
        for (int c = 0; c < 4; ++c) {
            const int k0 = c << 5;
            FragBF bf;
            bf.q[0] = *(const uint4*)&encT[brow + k0 + kh * 8];
            bf.q[1] = *(const uint4*)&encT[brow + k0 + 16 + kh * 8];
            acc = wmma_bf16(afr[c], bf, acc);
        }
#pragma unroll
        for (int r = 0; r < 8; ++r) {
            const int trow = mt * 16 + r + 8 * kh;
            out[((size_t)b * Ss + trow) * 1024 + Hh + hcol] = acc[r];
        }
    }
}

// ---------------------------------------------------------------------------
// Launch
// ---------------------------------------------------------------------------
extern "C" void kernel_launch(void* const* d_in, const int* in_sizes, int n_in,
                              void* d_out, int out_size, void* d_ws, size_t ws_size,
                              hipStream_t stream) {
    const int*   tok  = (const int*)d_in[0];
    const float* ehid = (const float*)d_in[1];   // [1][64][512]
    const float* enc  = (const float*)d_in[2];   // [64][128][512]
    const float* emb  = (const float*)d_in[3];   // [32000][512]
    const float* Wih  = (const float*)d_in[4];   // [1536][512]
    const float* Whh  = (const float*)d_in[5];   // [1536][512]
    const float* bih  = (const float*)d_in[6];   // [1536]
    const float* bhh  = (const float*)d_in[7];   // [1536]
    float* out = (float*)d_out;                  // [64][128][1024]

    char* ws = (char*)d_ws;
    unsigned short* WihB = (unsigned short*)(ws + OFF_WIH);
    unsigned short* WhhB = (unsigned short*)(ws + OFF_WHH);
    unsigned short* encB = (unsigned short*)(ws + OFF_ENC);
    unsigned short* encT = (unsigned short*)(ws + OFF_ENCT);
    float*          GI   = (float*)(ws + OFF_GI);
    unsigned short* Hbf  = (unsigned short*)(ws + OFF_HBF);

    prep_kernel<<<dim3(2048), dim3(256), 0, stream>>>(Wih, Whh, enc, WihB, WhhB, encB, encT);
    gi_gemm_kernel<<<dim3(512, 6), dim3(256), 0, stream>>>(tok, emb, WihB, bih, GI);
    recurrence_kernel<<<dim3(4), dim3(512), 0, stream>>>(ehid, WhhB, GI, bhh, Hbf, out);
    attention_kernel<<<dim3(8, 64), dim3(256), 0, stream>>>(Hbf, encB, encT, out);
}
